// ICGNN_42262478192808
// MI455X (gfx1250) — compile-verified, matrix operand order checked by reading, not run on hardware
//
#include <hip/hip_runtime.h>
#include <hip/hip_fp16.h>

// ---------------------------------------------------------------------------
// CDNA5 (gfx1250) implementation: line-graph GNN + JK-LSTM.
// - All matmuls: v_wmma_f32_16x16x32_f16 (wave32, 16x16 tiles, K chunks of 32)
// - Weights pre-converted f32->f16 once (zero-padded rows), msg buffers f16
// - Edge kernels stage the weight tile into LDS with global_load_async_to_lds
//   (+ s_wait_asynccnt), B fragments then come from LDS
// - Segment sums: global_atomic_add_f32 / global_atomic_pk_add_f16
// ---------------------------------------------------------------------------

typedef __attribute__((ext_vector_type(16))) _Float16 v16h;
typedef __attribute__((ext_vector_type(8)))  float    v8f;

#define EMB   64
#define FDIM  128
#define LG    64
#define KCAT  144   // 2*EMB + 16  (== 2*LG + 16)
#define KPAD  160   // KCAT padded up to a multiple of 32 (zero-filled)
#define HID   128
#define NJK   4
#define NCLS  40

__device__ __forceinline__ v8f wmma16(v16h a, v16h b, v8f c) {
  // (neg_a, A, neg_b, B, c_mod, C, reuse_a, reuse_b)
  return __builtin_amdgcn_wmma_f32_16x16x32_f16(false, a, false, b, (short)0, c,
                                                false, false);
}

// K index of f16-pair v (0..7) inside a 32-wide K chunk, per the ISA 16-bit
// A-matrix 16x32 layout: VGPR0..3 -> K={0..7}+8*half, VGPR4..7 -> K=16..23/+8
__device__ __forceinline__ int kmap8(int v, int half) {
  return ((v < 4) ? (v << 1) : (8 + (v << 1))) + (half << 3);
}

// Fragment from an f16 row (global or LDS), >= kbase+32 valid elements.
__device__ __forceinline__ v16h frag_f16(const _Float16* p, int half) {
  v16h f;
#pragma unroll
  for (int v = 0; v < 8; ++v) {
    int k = kmap8(v, half);
    f[2 * v]     = p[k];
    f[2 * v + 1] = p[k + 1];
  }
  return f;
}

// Fragment from a contiguous f32 row, converted to f16.
__device__ __forceinline__ v16h frag_f32(const float* __restrict__ p, int half) {
  v16h f;
#pragma unroll
  for (int v = 0; v < 8; ++v) {
    int k = kmap8(v, half);
    f[2 * v]     = (_Float16)p[k];
    f[2 * v + 1] = (_Float16)p[k + 1];
  }
  return f;
}

// Concat source: [s(64 f16) | d(64 f16) | attr(16 f32) | 0-pad]
__device__ __forceinline__ _Float16 cat144h(const _Float16* s, const _Float16* d,
                                            const float* __restrict__ a, int k) {
  if (k < 64)  return s[k];
  if (k < 128) return d[k - 64];
  if (k < 144) return (_Float16)a[k - 128];
  return (_Float16)0.f;
}

__device__ __forceinline__ v16h frag_cat144h(const _Float16* s, const _Float16* d,
                                             const float* __restrict__ a,
                                             int kbase, int half) {
  v16h f;
#pragma unroll
  for (int v = 0; v < 8; ++v) {
    int k = kbase + kmap8(v, half);
    f[2 * v]     = cat144h(s, d, a, k);
    f[2 * v + 1] = cat144h(s, d, a, k + 1);
  }
  return f;
}

// Stage `bytes` of f16 weights into LDS with per-lane async DMA (ASYNCcnt path).
__device__ __forceinline__ void stage_lds_async(const _Float16* g, _Float16* l,
                                                int bytes) {
  for (int off = threadIdx.x * 16; off < bytes; off += blockDim.x * 16) {
    unsigned laddr = (unsigned)(size_t)((char*)l + off);       // LDS byte offset
    const void* gp = (const char*)g + off;
    asm volatile("global_load_async_to_lds_b128 %0, %1, off"
                 :: "v"(laddr), "v"(gp) : "memory");
  }
  asm volatile("s_wait_asynccnt 0" ::: "memory");
}

// ---------------------------------------------------------------------------
// Pre-pass kernels: weight f32->f16 (with zero row padding), bias sum.
// ---------------------------------------------------------------------------
__global__ void k_cvt(const float* __restrict__ src, _Float16* __restrict__ dst,
                      int rows, int kdim, int kstride) {
  int i = blockIdx.x * blockDim.x + threadIdx.x;
  int tot = rows * kstride;
  if (i >= tot) return;
  int r = i / kstride, k = i - r * kstride;
  dst[i] = (_Float16)((k < kdim) ? src[(size_t)r * kdim + k] : 0.f);
}

__global__ void k_bias2(const float* __restrict__ a, const float* __restrict__ b,
                        float* __restrict__ o, int n) {
  int i = blockIdx.x * blockDim.x + threadIdx.x;
  if (i < n) o[i] = a[i] + b[i];
}

// ---------------------------------------------------------------------------
// Kernel 1: node_emb = relu(x @ W_feat^T) -> jk[0] (f32) and emb16 (f16 mirror)
// ---------------------------------------------------------------------------
__global__ __launch_bounds__(128) void k_feat(const float* __restrict__ x,
                                              const _Float16* __restrict__ Wf,
                                              float* __restrict__ out,
                                              _Float16* __restrict__ out16, int N) {
  int wave  = blockIdx.x * (blockDim.x >> 5) + (threadIdx.x >> 5);
  int tiles = (N + 15) >> 4;
  if (wave >= tiles) return;
  int lane = threadIdx.x & 31, half = lane >> 4, nloc = lane & 15;
  int m0 = wave << 4;
  int arow = m0 + nloc; if (arow >= N) arow = N - 1;

  v8f acc[4] = {};
#pragma unroll
  for (int kc = 0; kc < 4; ++kc) {
    v16h a = frag_f32(x + (size_t)arow * FDIM + kc * 32, half);
#pragma unroll
    for (int nt = 0; nt < 4; ++nt) {
      v16h b = frag_f16(Wf + (size_t)(nt * 16 + nloc) * FDIM + kc * 32, half);
      acc[nt] = wmma16(a, b, acc[nt]);
    }
  }
#pragma unroll
  for (int nt = 0; nt < 4; ++nt) {
#pragma unroll
    for (int r = 0; r < 8; ++r) {
      int m = m0 + half * 8 + r;
      if (m < N) {
        float v = acc[nt][r]; v = v > 0.f ? v : 0.f;
        out[(size_t)m * EMB + nt * 16 + nloc]   = v;
        out16[(size_t)m * EMB + nt * 16 + nloc] = (_Float16)v;
      }
    }
  }
}

// ---------------------------------------------------------------------------
// Kernel 2: msg_emb = relu([emb[src]|emb[dst]|attr] @ W_msg^T) -> msg (f16)
// W staged into LDS via async DMA.
// ---------------------------------------------------------------------------
__global__ __launch_bounds__(128) void k_msg_init(const _Float16* __restrict__ emb,
                                                  const int* __restrict__ srcI,
                                                  const int* __restrict__ dstI,
                                                  const float* __restrict__ attr,
                                                  const _Float16* __restrict__ W16,
                                                  _Float16* __restrict__ msg, int Eg) {
  __shared__ _Float16 Wl[64 * KPAD];   // 20 KB
  stage_lds_async(W16, Wl, 64 * KPAD * 2);
  __syncthreads();

  int wave  = blockIdx.x * (blockDim.x >> 5) + (threadIdx.x >> 5);
  int tiles = (Eg + 15) >> 4;
  if (wave >= tiles) return;
  int lane = threadIdx.x & 31, half = lane >> 4, nloc = lane & 15;
  int e0 = wave << 4;
  int erow = e0 + nloc; if (erow >= Eg) erow = Eg - 1;
  const _Float16* sp = emb + (size_t)srcI[erow] * EMB;
  const _Float16* dp = emb + (size_t)dstI[erow] * EMB;
  const float*    ap = attr + (size_t)erow * 16;

  v8f acc[4] = {};
#pragma unroll
  for (int kc = 0; kc < 5; ++kc) {
    v16h a = frag_cat144h(sp, dp, ap, kc * 32, half);
#pragma unroll
    for (int nt = 0; nt < 4; ++nt) {
      v16h b = frag_f16(&Wl[(nt * 16 + nloc) * KPAD + kc * 32], half);
      acc[nt] = wmma16(a, b, acc[nt]);
    }
  }
#pragma unroll
  for (int nt = 0; nt < 4; ++nt) {
#pragma unroll
    for (int r = 0; r < 8; ++r) {
      int e = e0 + half * 8 + r;
      if (e < Eg) {
        float v = acc[nt][r];
        msg[(size_t)e * LG + nt * 16 + nloc] = (_Float16)(v > 0.f ? v : 0.f);
      }
    }
  }
}

// ---------------------------------------------------------------------------
// Kernel 3: jk[1][dst[e]] += msg[e] @ W_lg2g^T + b  (bias per-edge == reference)
// ---------------------------------------------------------------------------
__global__ __launch_bounds__(128) void k_msgred(const _Float16* __restrict__ msg,
                                                const int* __restrict__ dstI,
                                                const _Float16* __restrict__ W16,
                                                const float* __restrict__ bias,
                                                float* __restrict__ nodeOut, int Eg) {
  int wave  = blockIdx.x * (blockDim.x >> 5) + (threadIdx.x >> 5);
  int tiles = (Eg + 15) >> 4;
  if (wave >= tiles) return;
  int lane = threadIdx.x & 31, half = lane >> 4, nloc = lane & 15;
  int e0 = wave << 4;
  int arow = e0 + nloc; if (arow >= Eg) arow = Eg - 1;

  v8f acc[4] = {};
#pragma unroll
  for (int kc = 0; kc < 2; ++kc) {
    v16h a = frag_f16(msg + (size_t)arow * LG + kc * 32, half);
#pragma unroll
    for (int nt = 0; nt < 4; ++nt) {
      v16h b = frag_f16(W16 + (size_t)(nt * 16 + nloc) * LG + kc * 32, half);
      acc[nt] = wmma16(a, b, acc[nt]);
    }
  }
  int dsts[8];
#pragma unroll
  for (int r = 0; r < 8; ++r) {
    int e = e0 + half * 8 + r;
    dsts[r] = (e < Eg) ? dstI[e] : -1;
  }
#pragma unroll
  for (int nt = 0; nt < 4; ++nt) {
    int col = nt * 16 + nloc;
    float bv = bias[col];
#pragma unroll
    for (int r = 0; r < 8; ++r)
      if (dsts[r] >= 0)
        atomicAdd(&nodeOut[(size_t)dsts[r] * EMB + col], acc[nt][r] + bv);
  }
}

// ---------------------------------------------------------------------------
// Kernel 4: line-graph layer:
//   msgOut[lgdst[e]] += relu([msg[s]|msg[d]|attr] @ Wl^T)   (pk f16 atomics)
// ---------------------------------------------------------------------------
__global__ __launch_bounds__(128) void k_lglayer(const _Float16* __restrict__ msg,
                                                 const int* __restrict__ srcI,
                                                 const int* __restrict__ dstI,
                                                 const float* __restrict__ attr,
                                                 const _Float16* __restrict__ W16,
                                                 _Float16* __restrict__ msgOut,
                                                 int Elg) {
  __shared__ _Float16 Wl[64 * KPAD];   // 20 KB
  stage_lds_async(W16, Wl, 64 * KPAD * 2);
  __syncthreads();

  int wave  = blockIdx.x * (blockDim.x >> 5) + (threadIdx.x >> 5);
  int tiles = (Elg + 15) >> 4;
  if (wave >= tiles) return;
  int lane = threadIdx.x & 31, half = lane >> 4, nloc = lane & 15;
  int e0 = wave << 4;
  int erow = e0 + nloc; if (erow >= Elg) erow = Elg - 1;
  const _Float16* sp = msg + (size_t)srcI[erow] * LG;
  const _Float16* dp = msg + (size_t)dstI[erow] * LG;
  const float*    ap = attr + (size_t)erow * 16;

  v8f acc[4] = {};
#pragma unroll
  for (int kc = 0; kc < 5; ++kc) {
    v16h a = frag_cat144h(sp, dp, ap, kc * 32, half);
#pragma unroll
    for (int nt = 0; nt < 4; ++nt) {
      v16h b = frag_f16(&Wl[(nt * 16 + nloc) * KPAD + kc * 32], half);
      acc[nt] = wmma16(a, b, acc[nt]);
    }
  }
  int dsts[8];
#pragma unroll
  for (int r = 0; r < 8; ++r) {
    int e = e0 + half * 8 + r;
    dsts[r] = (e < Elg) ? dstI[e] : -1;
  }
  bool even = (nloc & 1) == 0;
#pragma unroll
  for (int nt = 0; nt < 4; ++nt) {
    int col = nt * 16 + nloc;
    int colp = col & ~1;
#pragma unroll
    for (int r = 0; r < 8; ++r)
      if (dsts[r] >= 0) {
        float v = acc[nt][r]; v = v > 0.f ? v : 0.f;
        __half2 hv = __halves2half2(__float2half(even ? v : 0.f),
                                    __float2half(even ? 0.f : v));
        atomicAdd(reinterpret_cast<__half2*>(msgOut + (size_t)dsts[r] * LG + colp),
                  hv);   // global_atomic_pk_add_f16
      }
  }
}

// ---------------------------------------------------------------------------
// Kernel 5: elementwise scatter-add of f16 msg rows into f32 node accumulator.
// ---------------------------------------------------------------------------
__global__ void k_scatter64(const _Float16* __restrict__ src,
                            const int* __restrict__ dstI,
                            float* __restrict__ out, int Eg) {
  size_t tot = (size_t)Eg * 64;
  size_t stride = (size_t)gridDim.x * blockDim.x;
  for (size_t i = (size_t)blockIdx.x * blockDim.x + threadIdx.x; i < tot; i += stride) {
    size_t e = i >> 6; int c = (int)(i & 63);
    atomicAdd(&out[(size_t)dstI[e] * 64 + c], (float)src[i]);
  }
}

// ---------------------------------------------------------------------------
// Kernel 6: out = in @ W_lg2g^T + b   (64->64, per-node, direct store)
// ---------------------------------------------------------------------------
__global__ __launch_bounds__(128) void k_gemm64_bias(const float* __restrict__ in,
                                                     const _Float16* __restrict__ W16,
                                                     const float* __restrict__ bias,
                                                     float* __restrict__ out, int N) {
  int wave  = blockIdx.x * (blockDim.x >> 5) + (threadIdx.x >> 5);
  int tiles = (N + 15) >> 4;
  if (wave >= tiles) return;
  int lane = threadIdx.x & 31, half = lane >> 4, nloc = lane & 15;
  int m0 = wave << 4;
  int arow = m0 + nloc; if (arow >= N) arow = N - 1;

  v8f acc[4] = {};
#pragma unroll
  for (int kc = 0; kc < 2; ++kc) {
    v16h a = frag_f32(in + (size_t)arow * LG + kc * 32, half);
#pragma unroll
    for (int nt = 0; nt < 4; ++nt) {
      v16h b = frag_f16(W16 + (size_t)(nt * 16 + nloc) * LG + kc * 32, half);
      acc[nt] = wmma16(a, b, acc[nt]);
    }
  }
#pragma unroll
  for (int nt = 0; nt < 4; ++nt) {
    int col = nt * 16 + nloc;
    float bv = bias[col];
#pragma unroll
    for (int r = 0; r < 8; ++r) {
      int m = m0 + half * 8 + r;
      if (m < N) out[(size_t)m * EMB + col] = acc[nt][r] + bv;
    }
  }
}

// ---------------------------------------------------------------------------
// Kernel 7: bidirectional JK-LSTM + attention + output head + log-softmax.
// One wave per 16-node tile; gates in 16-col hid chunks (24 WMMAs each);
// h double-buffered f16 in LDS, c in LDS f32; attention score reduced via
// LDS float atomics (ds_add_f32) so h history is never materialized.
// ---------------------------------------------------------------------------
__global__ __launch_bounds__(32) void k_lstm(
    const float* __restrict__ jk, int N,
    const _Float16* __restrict__ Wih16_f, const _Float16* __restrict__ Whh16_f,
    const _Float16* __restrict__ Wih16_b, const _Float16* __restrict__ Whh16_b,
    const float* __restrict__ bsum_f, const float* __restrict__ bsum_b,
    const float* __restrict__ W_att, const float* __restrict__ b_att,
    const float* __restrict__ W_out, float* __restrict__ out) {
  __shared__ _Float16 xjk[NJK][16][EMB];   // 8 KB
  __shared__ _Float16 hbuf[2][16][HID];    // 8 KB (double-buffered h, f16)
  __shared__ float    cbuf[16][HID];       // 8 KB
  __shared__ float    score[16][NJK];

  int lane = threadIdx.x;          // one full wave32
  int half = lane >> 4, nloc = lane & 15;
  int m0 = blockIdx.x << 4;

  for (int i = lane; i < NJK * 16 * EMB; i += 32) {
    int t = i >> 10, rem = i & 1023, m = rem >> 6, c = rem & 63;
    int row = m0 + m; if (row >= N) row = N - 1;
    xjk[t][m][c] = (_Float16)jk[((size_t)t * N + row) * EMB + c];
  }
  if (lane < 16)
    for (int t = 0; t < NJK; ++t) score[lane][t] = 0.f;
  __syncthreads();

  for (int dir = 0; dir < 2; ++dir) {
    const _Float16* Wih = dir ? Wih16_b : Wih16_f;
    const _Float16* Whh = dir ? Whh16_b : Whh16_f;
    const float*    bsm = dir ? bsum_b : bsum_f;

    for (int i = lane; i < 16 * HID; i += 32) {
      (&hbuf[0][0][0])[i] = (_Float16)0.f;
      (&cbuf[0][0])[i] = 0.f;
    }
    __syncthreads();

    int cur = 0;
    for (int step = 0; step < NJK; ++step) {
      int t = dir ? (NJK - 1 - step) : step;
      for (int chunk = 0; chunk < 8; ++chunk) {
        int j0 = chunk * 16;
        v8f acc[4] = {};
        // x_t contribution (K = 64)
#pragma unroll
        for (int kc = 0; kc < 2; ++kc) {
          v16h a = frag_f16(&xjk[t][nloc][kc * 32], half);
#pragma unroll
          for (int g = 0; g < 4; ++g) {
            v16h b = frag_f16(Wih + (size_t)(g * HID + j0 + nloc) * EMB + kc * 32, half);
            acc[g] = wmma16(a, b, acc[g]);
          }
        }
        // h_{t-1} contribution (K = 128)
#pragma unroll
        for (int kc = 0; kc < 4; ++kc) {
          v16h a = frag_f16(&hbuf[cur][nloc][kc * 32], half);
#pragma unroll
          for (int g = 0; g < 4; ++g) {
            v16h b = frag_f16(Whh + (size_t)(g * HID + j0 + nloc) * HID + kc * 32, half);
            acc[g] = wmma16(a, b, acc[g]);
          }
        }
        int j = j0 + nloc;
        float bi = bsm[0 * HID + j], bf = bsm[1 * HID + j];
        float bg = bsm[2 * HID + j], bo = bsm[3 * HID + j];
        float watt = W_att[dir * HID + j];
#pragma unroll
        for (int r = 0; r < 8; ++r) {
          int m = half * 8 + r;
          float i_ = 1.f / (1.f + __expf(-(acc[0][r] + bi)));
          float f_ = 1.f / (1.f + __expf(-(acc[1][r] + bf)));
          float g_ = tanhf(acc[2][r] + bg);
          float o_ = 1.f / (1.f + __expf(-(acc[3][r] + bo)));
          float c_ = f_ * cbuf[m][j] + i_ * g_;
          cbuf[m][j] = c_;
          float h_ = o_ * tanhf(c_);
          hbuf[1 - cur][m][j] = (_Float16)h_;
          atomicAdd(&score[m][t], watt * h_);  // ds_add_f32
        }
      }
      __syncthreads();
      cur ^= 1;
    }
    __syncthreads();
  }

  if (lane < 16) {
    int row = m0 + lane;
    if (row < N) {
      float b0 = b_att[0];
      float s[NJK], mx = -1e30f;
#pragma unroll
      for (int t = 0; t < NJK; ++t) { s[t] = score[lane][t] + b0; mx = fmaxf(mx, s[t]); }
      float den = 0.f;
#pragma unroll
      for (int t = 0; t < NJK; ++t) { s[t] = __expf(s[t] - mx); den += s[t]; }
      float inv = 1.f / den;
      float logits[NCLS];
#pragma unroll
      for (int c2 = 0; c2 < NCLS; ++c2) logits[c2] = 0.f;
      for (int c = 0; c < EMB; ++c) {
        float fe = 0.f;
#pragma unroll
        for (int t = 0; t < NJK; ++t)
          fe += s[t] * inv * jk[((size_t)t * N + row) * EMB + c];
#pragma unroll
        for (int c2 = 0; c2 < NCLS; ++c2) logits[c2] += fe * W_out[c2 * EMB + c];
      }
      float mo = -1e30f;
#pragma unroll
      for (int c2 = 0; c2 < NCLS; ++c2) mo = fmaxf(mo, logits[c2]);
      float ds = 0.f;
#pragma unroll
      for (int c2 = 0; c2 < NCLS; ++c2) ds += __expf(logits[c2] - mo);
      float lse = mo + __logf(ds);
#pragma unroll
      for (int c2 = 0; c2 < NCLS; ++c2)
        out[(size_t)row * NCLS + c2] = logits[c2] - lse;
    }
  }
}

// ---------------------------------------------------------------------------
// Host orchestration (graph-capture safe: only async launches / memsets).
// ---------------------------------------------------------------------------
extern "C" void kernel_launch(void* const* d_in, const int* in_sizes, int n_in,
                              void* d_out, int out_size, void* d_ws, size_t ws_size,
                              hipStream_t stream) {
  const float* x      = (const float*)d_in[0];
  const int*   eig    = (const int*)d_in[1];    // [2][Eg]
  const float* eag    = (const float*)d_in[2];
  const int*   eilg   = (const int*)d_in[3];    // [2][Elg]
  const float* ealg   = (const float*)d_in[4];
  const float* W_feat = (const float*)d_in[5];
  const float* W_msg  = (const float*)d_in[6];
  const float* W_lg2g = (const float*)d_in[7];
  const float* b_lg2g = (const float*)d_in[8];
  const float* W_msgl = (const float*)d_in[9];  // [2][64][144]
  const float* Wih_f  = (const float*)d_in[10];
  const float* Whh_f  = (const float*)d_in[11];
  const float* bih_f  = (const float*)d_in[12];
  const float* bhh_f  = (const float*)d_in[13];
  const float* Wih_b  = (const float*)d_in[14];
  const float* Whh_b  = (const float*)d_in[15];
  const float* bih_b  = (const float*)d_in[16];
  const float* bhh_b  = (const float*)d_in[17];
  const float* W_att  = (const float*)d_in[18];
  const float* b_att  = (const float*)d_in[19];
  const float* W_out  = (const float*)d_in[20];
  float* out = (float*)d_out;

  const int N   = in_sizes[0] / FDIM;
  const int Eg  = in_sizes[1] / 2;
  const int Elg = in_sizes[3] / 2;

  // ---- workspace layout -------------------------------------------------
  char* p = (char*)d_ws;
  float* jk    = (float*)p;  p += (size_t)NJK * N * EMB * 4;
  float* tmpn  = (float*)p;  p += (size_t)N * EMB * 4;
  float* bsumf = (float*)p;  p += 512 * 4;
  float* bsumb = (float*)p;  p += 512 * 4;
  _Float16* msgA   = (_Float16*)p; p += (size_t)Eg * LG * 2;
  _Float16* msgB   = (_Float16*)p; p += (size_t)Eg * LG * 2;
  _Float16* emb16  = (_Float16*)p; p += (size_t)N * EMB * 2;
  _Float16* wf16   = (_Float16*)p; p += (size_t)64 * FDIM * 2;
  _Float16* wmsg16 = (_Float16*)p; p += (size_t)64 * KPAD * 2;
  _Float16* wlg16  = (_Float16*)p; p += (size_t)64 * LG * 2;
  _Float16* wl16   = (_Float16*)p; p += (size_t)128 * KPAD * 2;   // 2 layers
  _Float16* wihf16 = (_Float16*)p; p += (size_t)512 * EMB * 2;
  _Float16* whhf16 = (_Float16*)p; p += (size_t)512 * HID * 2;
  _Float16* wihb16 = (_Float16*)p; p += (size_t)512 * EMB * 2;
  _Float16* whhb16 = (_Float16*)p; p += (size_t)512 * HID * 2;

  auto cvt = [&](const float* s, _Float16* d, int rows, int kdim, int kstride) {
    int tot = rows * kstride;
    k_cvt<<<(tot + 255) / 256, 256, 0, stream>>>(s, d, rows, kdim, kstride);
  };
  auto tiles16 = [](int n) { return (n + 15) >> 4; };
  auto blocks4 = [&](int n) { return (tiles16(n) + 3) >> 2; };  // 4 waves / block

  // ---- weight pre-conversion (f32 -> f16, zero-padded rows) -------------
  cvt(W_feat, wf16,   64,  FDIM, FDIM);
  cvt(W_msg,  wmsg16, 64,  KCAT, KPAD);
  cvt(W_lg2g, wlg16,  64,  LG,   LG);
  cvt(W_msgl, wl16,   128, KCAT, KPAD);
  cvt(Wih_f,  wihf16, 512, EMB,  EMB);
  cvt(Whh_f,  whhf16, 512, HID,  HID);
  cvt(Wih_b,  wihb16, 512, EMB,  EMB);
  cvt(Whh_b,  whhb16, 512, HID,  HID);
  k_bias2<<<2, 256, 0, stream>>>(bih_f, bhh_f, bsumf, 512);
  k_bias2<<<2, 256, 0, stream>>>(bih_b, bhh_b, bsumb, 512);

  // 1) node embedding -> jk[0] (f32) + emb16 (f16)
  k_feat<<<blocks4(N), 128, 0, stream>>>(x, wf16, jk, emb16, N);

  // 2) initial edge-message features -> msgA (f16)
  k_msg_init<<<blocks4(Eg), 128, 0, stream>>>(emb16, eig, eig + Eg, eag, wmsg16,
                                              msgA, Eg);

  // 3) jk[1] = seg_sum(msgA @ W_lg2g^T + b, by dst)
  (void)hipMemsetAsync(jk + (size_t)N * EMB, 0, (size_t)N * EMB * sizeof(float), stream);
  k_msgred<<<blocks4(Eg), 128, 0, stream>>>(msgA, eig + Eg, wlg16, b_lg2g,
                                            jk + (size_t)N * EMB, Eg);

  // 4) two linegraph message-passing layers
  const _Float16* msrc = msgA;
  _Float16* mdst = msgB;
  for (int l = 0; l < 2; ++l) {
    (void)hipMemsetAsync(mdst, 0, (size_t)Eg * LG * sizeof(_Float16), stream);
    k_lglayer<<<blocks4(Elg), 128, 0, stream>>>(msrc, eilg, eilg + Elg, ealg,
                                                wl16 + (size_t)l * 64 * KPAD,
                                                mdst, Elg);
    (void)hipMemsetAsync(tmpn, 0, (size_t)N * EMB * sizeof(float), stream);
    k_scatter64<<<4096, 256, 0, stream>>>(mdst, eig + Eg, tmpn, Eg);
    k_gemm64_bias<<<blocks4(N), 128, 0, stream>>>(tmpn, wlg16, b_lg2g,
                                                  jk + (size_t)(l + 2) * N * EMB, N);
    const _Float16* sw = msrc; msrc = mdst; mdst = (_Float16*)sw;
  }

  // 5) JK bidirectional LSTM + attention + output head + log-softmax
  k_lstm<<<tiles16(N), 32, 0, stream>>>(jk, N, wihf16, whhf16, wihb16, whhb16,
                                        bsumf, bsumb, W_att, b_att, W_out, out);
}
